// RotaryLinearAttention_13134009991504
// MI455X (gfx1250) — compile-verified, matrix operand order checked by reading, not run on hardware
//
#include <hip/hip_runtime.h>
#include <hip/hip_bf16.h>

// ---------------------------------------------------------------------------
// RotaryLinearAttention forward for gfx1250 (MI455X, wave32, WMMA + TDM)
//   DIM=512, N_HEADS=8, N_KV_HEADS=4, HEAD_DIM=64, B=4, S=8192 (tokens=32768)
// ---------------------------------------------------------------------------

typedef __attribute__((ext_vector_type(16))) _Float16 v16h;
typedef __attribute__((ext_vector_type(8)))  _Float16 v8h;
typedef __attribute__((ext_vector_type(8)))  float    v8f;
typedef __attribute__((ext_vector_type(4)))  unsigned int u32x4;
typedef __attribute__((ext_vector_type(8)))  int          i32x8;
typedef __attribute__((ext_vector_type(4)))  int          i32x4;

#define TOKENS   32768
#define SEQ      8192
#define DIM      512
#define NH       8
#define NKV      4
#define HD       64
#define KDIM     512

union F16Frag { v16h v; v8h h[2]; };

// Fragment load for the 16-bit A/B layout of V_WMMA_F32_16X16X32_F16:
// lane group kg (= lane>>4) holds halves K = kg*8.. (slots 0..7) and
// K = 16+kg*8.. (slots 8..15).  Two 16-byte loads (global_load_b128 or
// ds_load_b128 depending on the pointer's address space).
static __device__ __forceinline__ v16h load_frag(const _Float16* __restrict__ base, int kg) {
    F16Frag f;
    f.h[0] = *(const v8h*)(base + kg * 8);
    f.h[1] = *(const v8h*)(base + 16 + kg * 8);
    return f.v;
}

#if __has_builtin(__builtin_amdgcn_tensor_load_to_lds)
// Issue one TDM transfer: copy a (64 rows x 512 halves) panel of B
// (row-major, row stride 512 halves) into LDS at byte offset lds_byte_off.
// D# built per CDNA5 ISA 8.3/8.4: group0 = {count/flags, lds_addr,
// global_addr[56:0], type=2}; group1 = {data_size=2B, tensor_dim0=512,
// tensor_dim1=nrows_total, tile_dim0=512, tile_dim1=64, dim0_stride=512};
// groups 2/3 (and the trailing group) zero for a 2-D tile.
// clang-23 / therock-10.0 builtin arity: 6 args
//   (uint32x4 g0, int32x8 g1, int32x4 g2, int32x4 g3, int32x8 g4, i32 cpol)
static __device__ __forceinline__ void tdm_load_panel(
    const _Float16* gsrc, unsigned int lds_byte_off, int nrows_total)
{
    unsigned long long ga = (unsigned long long)(size_t)gsrc;
    u32x4 g0;
    g0[0] = 1u;                                                // count=1 (valid user D#)
    g0[1] = lds_byte_off;                                      // lds_addr (bytes)
    g0[2] = (unsigned int)(ga & 0xffffffffull);                // global_addr[31:0]
    g0[3] = (unsigned int)((ga >> 32) & 0x01ffffffull)         // global_addr[56:32]
          | 0x80000000u;                                       // type=2 ("image")
    i32x8 g1;
    g1[0] = 0x00010000;                                        // data_size code 1 = 2 bytes
    g1[1] = (int)(512u << 16);                                 // tensor_dim0 = 512
    g1[2] = (int)(((unsigned int)nrows_total & 0xffffu) << 16);// tensor_dim1 lo16
    g1[3] = (int)(512u << 16);                                 // tile_dim0 = 512
    g1[4] = 64;                                                // tile_dim1 = 64 rows
    g1[5] = 512;                                               // tensor_dim0_stride = 512
    g1[6] = 0;
    g1[7] = 0;
    i32x4 gz4 = {0, 0, 0, 0};
    i32x8 gz8 = {0, 0, 0, 0, 0, 0, 0, 0};
    __builtin_amdgcn_tensor_load_to_lds(g0, g1, gz4, gz4, gz8, 0);
    __builtin_amdgcn_s_wait_tensorcnt(0);
}
#endif

// ---------------------------------------------------------------------------
// NT GEMM: C[M,N] f32 = A[M,512] f16 * B[N,512]^T f16 (both K-contiguous).
// Block = 256 threads = 8 waves; block tile 256(M) x 64(N); wave tile 32x64.
// B panel (64 x 512 f16 = 64 KB) staged in LDS by the Tensor Data Mover and
// reused by all 8 waves across all 16 k-steps; A streams via global_load_b128.
// grid = (N/64, M/256).
// ---------------------------------------------------------------------------
__global__ __launch_bounds__(256) void gemm_nt_f16(
    const _Float16* __restrict__ A, const _Float16* __restrict__ B,
    float* __restrict__ C, int N)
{
    __shared__ __align__(16) _Float16 Bs[64 * KDIM];   // 64 KB

    const int lane = threadIdx.x & 31;
    const int wave = threadIdx.x >> 5;
    const int rowBase = blockIdx.y * 256 + wave * 32;
    const int colBase = blockIdx.x * 64;
    const int mlane = lane & 15;
    const int kg    = lane >> 4;

#if __has_builtin(__builtin_amdgcn_tensor_load_to_lds)
    if (wave == 0)
        tdm_load_panel(B + (size_t)colBase * KDIM,
                       (unsigned int)(size_t)(&Bs[0]), N);
#else
    for (int i = threadIdx.x; i < 64 * KDIM / 8; i += 256)
        ((v8h*)Bs)[i] = ((const v8h*)(B + (size_t)colBase * KDIM))[i];
#endif
    __syncthreads();

    const _Float16* arow0 = A + (size_t)(rowBase + mlane) * KDIM;
    const _Float16* arow1 = arow0 + (size_t)16 * KDIM;
    const _Float16* brow  = Bs + (size_t)mlane * KDIM;

    v8f acc[2][4];
    #pragma unroll
    for (int r = 0; r < 2; ++r)
        #pragma unroll
        for (int j = 0; j < 4; ++j)
            acc[r][j] = (v8f)(0.0f);

    for (int kk = 0; kk < KDIM; kk += 32) {
        v16h a0 = load_frag(arow0 + kk, kg);
        v16h a1 = load_frag(arow1 + kk, kg);
        #pragma unroll
        for (int j = 0; j < 4; ++j) {
            v16h b = load_frag(brow + (size_t)(j * 16) * KDIM + kk, kg);
            acc[0][j] = __builtin_amdgcn_wmma_f32_16x16x32_f16(
                false, a0, false, b, (short)0, acc[0][j], false, false);
            acc[1][j] = __builtin_amdgcn_wmma_f32_16x16x32_f16(
                false, a1, false, b, (short)0, acc[1][j], false, false);
        }
    }

    // D layout: VGPR v, lanes 0-15 -> M=v, N=lane; lanes 16-31 -> M=v+8, N=lane-16
    const int nCol  = colBase + (lane & 15);
    const int mHalf = (lane >> 4) * 8;
    #pragma unroll
    for (int r = 0; r < 2; ++r)
        #pragma unroll
        for (int j = 0; j < 4; ++j)
            #pragma unroll
            for (int v = 0; v < 8; ++v) {
                int m = rowBase + r * 16 + mHalf + v;
                C[(size_t)m * N + nCol + j * 16] = acc[r][j][v];
            }
}

// ---------------------------------------------------------------------------
// RMSNorm over DIM=512 + cast to f16.  One block per token, 256 threads.
// ---------------------------------------------------------------------------
__global__ __launch_bounds__(256) void rmsnorm_cast(
    const float* __restrict__ x, const float* __restrict__ w,
    _Float16* __restrict__ xh)
{
    __shared__ float red[8];
    const size_t t = blockIdx.x;
    const float* xr = x + t * DIM;
    const int tid = threadIdx.x;
    float v0 = xr[tid], v1 = xr[tid + 256];
    float ss = v0 * v0 + v1 * v1;
    #pragma unroll
    for (int o = 16; o > 0; o >>= 1) ss += __shfl_xor(ss, o, 32);
    if ((tid & 31) == 0) red[tid >> 5] = ss;
    __syncthreads();
    float sum = 0.0f;
    #pragma unroll
    for (int i = 0; i < 8; ++i) sum += red[i];
    const float r = rsqrtf(sum * (1.0f / DIM) + 1e-6f);
    xh[t * DIM + tid]       = (_Float16)(v0 * r * w[tid]);
    xh[t * DIM + tid + 256] = (_Float16)(v1 * r * w[tid + 256]);
}

// ---------------------------------------------------------------------------
// Pack wq(512x512) / wk(256x512) / wv(256x512) into one f16 panel (1024x512),
// and wo(512x512) -> f16.
// ---------------------------------------------------------------------------
__global__ __launch_bounds__(256) void cast_weights(
    const float* __restrict__ wq, const float* __restrict__ wk,
    const float* __restrict__ wv, const float* __restrict__ wo,
    _Float16* __restrict__ wqkv, _Float16* __restrict__ woh)
{
    const int i = blockIdx.x * 256 + threadIdx.x;     // 0 .. 1024*512-1
    const int QW = 512 * 512, KW = 256 * 512;
    if (i < QW)                wqkv[i] = (_Float16)wq[i];
    else if (i < QW + KW)      wqkv[i] = (_Float16)wk[i - QW];
    else                       wqkv[i] = (_Float16)wv[i - QW - KW];
    if (i < QW)                woh[i]  = (_Float16)wo[i];
}

// ---------------------------------------------------------------------------
// In-place RoPE + phi (elu+1) on Q (cols 0..511) and K (cols 512..767) of the
// fused QKV f32 buffer (row stride 1024).  One block per token.
// ---------------------------------------------------------------------------
static __device__ __forceinline__ float phi_elu1(float x) {
    return x > 0.0f ? x + 1.0f : __expf(x);
}

__global__ __launch_bounds__(256) void rope_phi_qk(float* __restrict__ qkv)
{
    const size_t t = blockIdx.x;
    const int s = (int)(t & (SEQ - 1));
    float* row = qkv + t * 1024;
    const int tid = threadIdx.x;
    const int d2  = tid & 31;
    const float inv = __powf(10000.0f, -(float)d2 * (1.0f / 32.0f));
    const float ang = (float)s * inv;
    float sn, cs;
    __sincosf(ang, &sn, &cs);
    // Q pair: col = 2*tid  (head = tid/32, d2 = tid%32)
    {
        float xr = row[2 * tid], xi = row[2 * tid + 1];
        row[2 * tid]     = phi_elu1(xr * cs - xi * sn);
        row[2 * tid + 1] = phi_elu1(xr * sn + xi * cs);
    }
    if (tid < 128) { // K pair: col = 512 + 2*tid
        float xr = row[512 + 2 * tid], xi = row[512 + 2 * tid + 1];
        row[512 + 2 * tid]     = phi_elu1(xr * cs - xi * sn);
        row[512 + 2 * tid + 1] = phi_elu1(xr * sn + xi * cs);
    }
}

// ---------------------------------------------------------------------------
// KV[b,h,d]  = sum_s phiK[b,s,h,d] * V[b,s,h,d]
// KS[b,h,d]  = sum_s phiK[b,s,h,d]            (per kv-head, d<64)
// grid = 4(b) * 4(h) * 16(chunks of 512 positions); block = 256.
// ---------------------------------------------------------------------------
__global__ __launch_bounds__(256) void kv_reduce(
    const float* __restrict__ qkv, float* __restrict__ kvacc,
    float* __restrict__ ksacc)
{
    const int chunk = blockIdx.x & 15;
    const int h     = (blockIdx.x >> 4) & 3;
    const int b     = blockIdx.x >> 6;
    const int d     = threadIdx.x & 63;
    const int sr    = threadIdx.x >> 6;        // 0..3
    size_t base = ((size_t)b * SEQ + (size_t)chunk * 512 + sr) * 1024
                + 512 + h * HD + d;
    float kv = 0.0f, ks = 0.0f;
    for (int i = 0; i < 128; ++i) {
        const float* p = qkv + base + (size_t)i * 4 * 1024;
        float pk = p[0];        // phiK
        float vv = p[256];      // V (cols 768.. = K cols + 256)
        kv += pk * vv;
        ks += pk;
    }
    __shared__ float skv[256], sks[256];
    skv[threadIdx.x] = kv; sks[threadIdx.x] = ks;
    __syncthreads();
    if (threadIdx.x < 64) {
        float tkv = skv[d] + skv[d + 64] + skv[d + 128] + skv[d + 192];
        float tks = sks[d] + sks[d + 64] + sks[d + 128] + sks[d + 192];
        atomicAdd(&kvacc[(b * NKV + h) * HD + d], tkv);
        atomicAdd(&ksacc[(b * NKV + h) * HD + d], tks);
    }
}

__global__ __launch_bounds__(256) void zero_acc(float* __restrict__ p, int n)
{
    for (int i = blockIdx.x * 256 + threadIdx.x; i < n; i += gridDim.x * 256)
        p[i] = 0.0f;
}

// ---------------------------------------------------------------------------
// Y[t, h*64+d] = phiQ * KV[b,h/2,d] / (phiQ . KS[b,h/2,:] + eps) -> f16
// One block per token; wave w handles head w (wave32 dot reduction).
// ---------------------------------------------------------------------------
__global__ __launch_bounds__(256) void compute_y(
    const float* __restrict__ qkv, const float* __restrict__ kvacc,
    const float* __restrict__ ksacc, _Float16* __restrict__ yh)
{
    const size_t t = blockIdx.x;
    const int b    = (int)(t >> 13);
    const int h    = threadIdx.x >> 5;
    const int lane = threadIdx.x & 31;
    const int kvh  = h >> 1;
    const float* q  = qkv + t * 1024 + h * HD;
    const float* KV = kvacc + (b * NKV + kvh) * HD;
    const float* KS = ksacc + (b * NKV + kvh) * HD;
    float q0 = q[lane], q1 = q[lane + 32];
    float z = q0 * KS[lane] + q1 * KS[lane + 32];
    #pragma unroll
    for (int o = 16; o > 0; o >>= 1) z += __shfl_xor(z, o, 32);
    const float inv = 1.0f / (z + 1e-6f);
    _Float16* y = yh + t * DIM + h * HD;
    y[lane]      = (_Float16)(q0 * KV[lane] * inv);
    y[lane + 32] = (_Float16)(q1 * KV[lane + 32] * inv);
}

// ---------------------------------------------------------------------------
extern "C" void kernel_launch(void* const* d_in, const int* in_sizes, int n_in,
                              void* d_out, int out_size, void* d_ws, size_t ws_size,
                              hipStream_t stream)
{
    const float* x      = (const float*)d_in[0];
    const float* norm_w = (const float*)d_in[1];
    const float* wq     = (const float*)d_in[2];
    const float* wk     = (const float*)d_in[3];
    const float* wv     = (const float*)d_in[4];
    const float* wo     = (const float*)d_in[5];
    float* out          = (float*)d_out;

    char* p = (char*)d_ws;
    _Float16* xh   = (_Float16*)p;  p += (size_t)TOKENS * DIM * 2;       // 32 MB
    _Float16* wqkv = (_Float16*)p;  p += (size_t)1024 * DIM * 2;         // 1 MB
    _Float16* woh  = (_Float16*)p;  p += (size_t)DIM * DIM * 2;          // 0.5 MB
    float*    qkv  = (float*)p;     p += (size_t)TOKENS * 1024 * 4;      // 128 MB
    _Float16* yh   = (_Float16*)p;  p += (size_t)TOKENS * DIM * 2;       // 32 MB
    float*    kvacc = (float*)p;    p += 4 * NKV * HD * 4;
    float*    ksacc = (float*)p;    p += 4 * NKV * HD * 4;

    // 1) RMSNorm + f16 cast
    rmsnorm_cast<<<TOKENS, 256, 0, stream>>>(x, norm_w, xh);

    // 2) weight casts (1024*512 elems -> 2048 blocks)
    cast_weights<<<(1024 * 512) / 256, 256, 0, stream>>>(wq, wk, wv, wo, wqkv, woh);

    // 3) fused QKV GEMM: (32768 x 512) * (1024 x 512)^T -> f32 (32768 x 1024)
    {
        dim3 grid(1024 / 64, TOKENS / 256);
        gemm_nt_f16<<<grid, 256, 0, stream>>>(xh, wqkv, qkv, 1024);
    }

    // 4) RoPE + phi on Q and K regions (in place)
    rope_phi_qk<<<TOKENS, 256, 0, stream>>>(qkv);

    // 5) KV / K_sum reductions
    zero_acc<<<1, 256, 0, stream>>>(kvacc, 2 * 4 * NKV * HD); // zeros both (contiguous)
    kv_reduce<<<4 * NKV * 16, 256, 0, stream>>>(qkv, kvacc, ksacc);

    // 6) Y = phiQ * KV / (phiQ . KS + eps), cast to f16
    compute_y<<<TOKENS, 256, 0, stream>>>(qkv, kvacc, ksacc, yh);

    // 7) output GEMM: (32768 x 512) * (512 x 512)^T -> d_out f32
    {
        dim3 grid(DIM / 64, TOKENS / 256);
        gemm_nt_f16<<<grid, 256, 0, stream>>>(yh, woh, out, DIM);
    }
}
